// BasicBlock_73933567033945
// MI455X (gfx1250) — matrix operand order repaired and checked by reading, not compiled, not run on hardware
//
#include <hip/hip_runtime.h>

// ---------------------------------------------------------------------------
// Problem constants (from the reference)
// ---------------------------------------------------------------------------
#define C_DIM   128
#define WIDTH_D 32
#define KC_D    24
#define SEQ_L   11
#define RADIUS  12.0f
#define NEG_IN  0.1f
#define NEG_K   0.2f

typedef __attribute__((ext_vector_type(16))) _Float16 v16h;
typedef __attribute__((ext_vector_type(8)))  _Float16 v8h;
typedef __attribute__((ext_vector_type(4)))  _Float16 v4h;
typedef __attribute__((ext_vector_type(8)))  float    v8f;

__device__ __forceinline__ float lrelu_f(float v, float a) {
    return v >= 0.0f ? v : a * v;
}

__device__ __forceinline__ v8f wmma_f16(v16h a, v16h b, v8f c) {
    // D(16x16 f32) = A(16x32 f16) * B(32x16 f16) + C
    return __builtin_amdgcn_wmma_f32_16x16x32_f16(
        false, a, false, b, (short)0, c, false, false);
}

// --- CDNA5 async copy: global -> LDS, 16 bytes per lane (ASYNCcnt) ----------
__device__ __forceinline__ void async_copy_b128(void* lds_dst, const void* gsrc) {
    unsigned lds = (unsigned)(unsigned long long)lds_dst;   // addr[31:0] = LDS offset
    asm volatile("global_load_async_to_lds_b128 %0, %1, off"
                 :: "v"(lds), "v"(gsrc)
                 : "memory");
}
__device__ __forceinline__ void wait_async() {
    asm volatile("s_wait_asynccnt 0x0" ::: "memory");
}

// ---------------------------------------------------------------------------
// WMMA operand helpers (CDNA5 ISA 7.12.2, wave32)
// ---------------------------------------------------------------------------
// A 16x32 f16 tile from row-major f16 memory: lane (half = lane>>4, m = lane&15)
// holds two contiguous 8-half runs: K = half*8..+7 and K = 16+half*8..+7.
__device__ __forceinline__ v16h load_a_rowmajor(const _Float16* __restrict__ base,
                                                int row0, int strideH, int kc) {
    const int lane = threadIdx.x & 31;
    const _Float16* p = base + (size_t)(row0 + (lane & 15)) * strideH
                             + kc + (lane >> 4) * 8;
    v8h lo = *(const v8h*)p;          // K = half*8 .. +7
    v8h hi = *(const v8h*)(p + 16);   // K = 16+half*8 .. +7
    return __builtin_shufflevector(lo, hi, 0, 1, 2, 3, 4, 5, 6, 7,
                                           8, 9, 10, 11, 12, 13, 14, 15);
}

// B 32x16 f16 tile from the pre-packed "B-ready" buffer: lane's 16 halfs are
// contiguous: bready[(tileIdx*32 + lane)*16 .. +15].
__device__ __forceinline__ v16h load_b_ready(const _Float16* __restrict__ bready,
                                             int tileIdx) {
    const int lane = threadIdx.x & 31;
    return *(const v16h*)(bready + ((size_t)tileIdx * 32 + lane) * 16);
}

// Generic element-wise builders (used only for the LDS-resident k3 tiles).
template <typename F>
__device__ __forceinline__ v16h make_a_tile(F getv) {
    const int lane = threadIdx.x & 31;
    const int half = lane >> 4;
    const int m    = lane & 15;
    v16h a;
#pragma unroll
    for (int v = 0; v < 8; ++v) {
        const int kb = (v < 4) ? (2 * v + half * 8) : (16 + 2 * (v - 4) + half * 8);
        a[2 * v]     = getv(m, kb);
        a[2 * v + 1] = getv(m, kb + 1);
    }
    return a;
}
template <typename F>
__device__ __forceinline__ v16h make_b_tile(F getv) {
    const int lane = threadIdx.x & 31;
    const int half = lane >> 4;
    const int n    = lane & 15;
    v16h b;
#pragma unroll
    for (int v = 0; v < 8; ++v) {
        const int k0 = (v < 4) ? (2 * v + half * 8) : (16 + 2 * (v - 4) + half * 8);
        b[2 * v]     = getv(k0, n);
        b[2 * v + 1] = getv(k0 + 1, n);
    }
    return b;
}
// C/D 16x16 f32: VGPR r -> M = r + 8*half, N = lane&15.
template <typename F>
__device__ __forceinline__ void store_d(v8f d, F putv) {
    const int lane = threadIdx.x & 31;
    const int half = lane >> 4;
    const int n    = lane & 15;
#pragma unroll
    for (int r = 0; r < 8; ++r) putv(r + half * 8, n, d[r]);
}

// ---------------------------------------------------------------------------
// k0a: xr = f16(lrelu(x, .1)) — vectorized f32x4 -> f16x4
// ---------------------------------------------------------------------------
__global__ __launch_bounds__(256)
void k0_lrelu_x(const float* __restrict__ x, _Float16* __restrict__ xr, int total4) {
    const int i = blockIdx.x * blockDim.x + threadIdx.x;
    if (i >= total4) return;
    const float4 v = ((const float4*)x)[i];
    v4h o;
    o[0] = (_Float16)lrelu_f(v.x, NEG_IN);
    o[1] = (_Float16)lrelu_f(v.y, NEG_IN);
    o[2] = (_Float16)lrelu_f(v.z, NEG_IN);
    o[3] = (_Float16)lrelu_f(v.w, NEG_IN);
    ((v4h*)xr)[i] = o;
}

// ---------------------------------------------------------------------------
// k0b: pack a row-major f32 weight [Ktot, Nw] into B-tile-ready f16 layout.
// blockIdx.x = chunk*(Nw/16) + t ; lane writes its 16 operand halfs contiguously.
// ---------------------------------------------------------------------------
__global__ __launch_bounds__(32)
void k0_pack_b(const float* __restrict__ W, _Float16* __restrict__ out, int Nw) {
    const int NT    = Nw / 16;
    const int chunk = blockIdx.x / NT;
    const int t     = blockIdx.x % NT;
    const int lane  = threadIdx.x;
    const int half  = lane >> 4, n = lane & 15;
    _Float16* o = out + ((size_t)blockIdx.x * 32 + lane) * 16;
#pragma unroll
    for (int v = 0; v < 8; ++v) {
        const int k = chunk * 32 + ((v < 4) ? (2 * v + half * 8) : (16 + 2 * (v - 4) + half * 8));
        o[2 * v]     = (_Float16)W[(size_t)k * Nw + t * 16 + n];
        o[2 * v + 1] = (_Float16)W[(size_t)(k + 1) * Nw + t * 16 + n];
    }
}

// ---------------------------------------------------------------------------
// k1: h = f16(lrelu(xr @ W_in, .1))   [N,128] x [128,32]
// ---------------------------------------------------------------------------
__global__ __launch_bounds__(32)
void k1_input_mlp(const _Float16* __restrict__ xr, const _Float16* __restrict__ bWin,
                  _Float16* __restrict__ h) {
    const int row0 = blockIdx.x * 16;
    v8f acc0 = {}; v8f acc1 = {};
#pragma unroll
    for (int c = 0; c < C_DIM / 32; ++c) {
        v16h a  = load_a_rowmajor(xr, row0, C_DIM, c * 32);
        v16h b0 = load_b_ready(bWin, c * 2 + 0);
        v16h b1 = load_b_ready(bWin, c * 2 + 1);
        acc0 = wmma_f16(a, b0, acc0);
        acc1 = wmma_f16(a, b1, acc1);
    }
    store_d(acc0, [&](int m, int n, float v) {
        h[(size_t)(row0 + m) * WIDTH_D + n] = (_Float16)lrelu_f(v, NEG_IN);
    });
    store_d(acc1, [&](int m, int n, float v) {
        h[(size_t)(row0 + m) * WIDTH_D + 16 + n] = (_Float16)lrelu_f(v, NEG_IN);
    });
}

// ---------------------------------------------------------------------------
// k2: per-edge continuous kernel kern[E,24] (f16), VALU + vectorized store.
// ---------------------------------------------------------------------------
__global__ __launch_bounds__(256)
void k2_edge_kern(const float* __restrict__ pos, const float* __restrict__ seq,
                  const float* __restrict__ ori, const float* __restrict__ Ws0,
                  const float* __restrict__ bs0, const int* __restrict__ src,
                  const int* __restrict__ dst, int E,
                  _Float16* __restrict__ kern) {
    const int e = blockIdx.x * blockDim.x + threadIdx.x;
    if (e >= E) return;
    const int i = dst[e];
    const int j = src[e];

    const float dx = pos[j * 3 + 0] - pos[i * 3 + 0];
    const float dy = pos[j * 3 + 1] - pos[i * 3 + 1];
    const float dz = pos[j * 3 + 2] - pos[i * 3 + 2];
    const float dist = sqrtf(dx * dx + dy * dy + dz * dz);
    const float inv  = 1.0f / (dist + 1e-9f);
    const float dir[3] = {dx * inv, dy * inv, dz * inv};

    float delta[7];
#pragma unroll
    for (int r = 0; r < 3; ++r) {
        float loc = 0.0f, of = 0.0f;
#pragma unroll
        for (int c = 0; c < 3; ++c) {
            const float ri = ori[(size_t)i * 9 + r * 3 + c];
            loc += ri * dir[c];
            of  += ri * ori[(size_t)j * 9 + r * 3 + c];
        }
        delta[r]     = loc;
        delta[3 + r] = of;
    }
    delta[6] = dist / RADIUS;

    const float s_half = (float)(SEQ_L / 2);
    float sdiff = seq[j] - seq[i];
    sdiff = fminf(fmaxf(sdiff, -s_half), s_half);
    const int sidx = (int)(sdiff + s_half);

    const float* W  = Ws0 + (size_t)sidx * 7 * KC_D;
    const float* bp = bs0 + (size_t)sidx * KC_D;
    float vals[KC_D];
#pragma unroll
    for (int k = 0; k < KC_D; ++k) {
        float s = bp[k];
#pragma unroll
        for (int d = 0; d < 7; ++d) s += delta[d] * W[d * KC_D + k];
        vals[k] = lrelu_f(s, NEG_K);
    }
    v4h* krow = (v4h*)(kern + (size_t)e * KC_D);
#pragma unroll
    for (int g = 0; g < KC_D / 4; ++g) {
        v4h o;
#pragma unroll
        for (int q = 0; q < 4; ++q) o[q] = (_Float16)vals[g * 4 + q];
        krow[g] = o;
    }
}

// ---------------------------------------------------------------------------
// k3: per-node aggregation  agg[n] = kern_nᵀ(24x17) @ h_nb(17x32)
// One wave per node; async-LDS staging + 4x wmma (padded 32x32x32).
// ---------------------------------------------------------------------------
__global__ __launch_bounds__(32)
void k3_aggregate(const _Float16* __restrict__ h,
                  const _Float16* __restrict__ kern,
                  const int* __restrict__ src, int epn,
                  _Float16* __restrict__ agg) {
    const int n    = blockIdx.x;
    const int lane = threadIdx.x;
    const int eb   = n * epn;

    __shared__ int s_src[32];
    __shared__ __align__(16) _Float16 s_h[32 * WIDTH_D];     // [k][w]
    __shared__ __align__(16) _Float16 s_k[32 * KC_D + 8];    // [k][m]

    // (a) kern block for this node is contiguous: async DMA to LDS.
    {
        const char* g = (const char*)(kern + (size_t)eb * KC_D);
        const int nxfer = (epn * KC_D * 2 + 15) / 16;        // 51 for epn=17
        for (int i = lane; i < nxfer; i += 32)
            async_copy_b128(&s_k[i * 8], g + i * 16);
    }
    // (b) neighbor indices (plain loads).
    if (lane < epn) s_src[lane] = src[eb + lane];
    __syncthreads();
    // (c) gather neighbor h rows (64B each) via async DMA to LDS.
    for (int i = lane; i < epn * 4; i += 32) {
        const int row = i >> 2, q = i & 3;
        async_copy_b128(&s_h[row * WIDTH_D + q * 8],
                        h + (size_t)s_src[row] * WIDTH_D + q * 8);
    }
    wait_async();
    __syncthreads();

    const _Float16 hzero = (_Float16)0.0f;
    // A[m][k] = kernᵀ (padded); B[k][w] = neighbor features (padded).
    v16h a0 = make_a_tile([&](int m, int k) {
        return (k < epn && m < KC_D) ? s_k[k * KC_D + m] : hzero;
    });
    v16h a1 = make_a_tile([&](int m, int k) {
        const int mg = m + 16;
        return (k < epn && mg < KC_D) ? s_k[k * KC_D + mg] : hzero;
    });
    v16h b0 = make_b_tile([&](int k, int w) {
        return (k < epn) ? s_h[k * WIDTH_D + w] : hzero;
    });
    v16h b1 = make_b_tile([&](int k, int w) {
        return (k < epn) ? s_h[k * WIDTH_D + 16 + w] : hzero;
    });

    v8f z = {};
    v8f d00 = wmma_f16(a0, b0, z);
    v8f d01 = wmma_f16(a0, b1, z);
    v8f d10 = wmma_f16(a1, b0, z);
    v8f d11 = wmma_f16(a1, b1, z);

    _Float16* arow = agg + (size_t)n * (KC_D * WIDTH_D);
    store_d(d00, [&](int m, int w, float v) {
        if (m < KC_D) arow[m * WIDTH_D + w] = (_Float16)v;
    });
    store_d(d01, [&](int m, int w, float v) {
        if (m < KC_D) arow[m * WIDTH_D + 16 + w] = (_Float16)v;
    });
    store_d(d10, [&](int m, int w, float v) {
        if (m + 16 < KC_D) arow[(m + 16) * WIDTH_D + w] = (_Float16)v;
    });
    store_d(d11, [&](int m, int w, float v) {
        if (m + 16 < KC_D) arow[(m + 16) * WIDTH_D + 16 + w] = (_Float16)v;
    });
}

// ---------------------------------------------------------------------------
// k4: conv = agg(N,768) @ Wk(768,32); convL = f16(lrelu(conv, .1))
// ---------------------------------------------------------------------------
__global__ __launch_bounds__(32)
void k4_conv_gemm(const _Float16* __restrict__ agg, const _Float16* __restrict__ bWk,
                  _Float16* __restrict__ convL) {
    const int row0 = blockIdx.x * 16;
    const int KTOT = KC_D * WIDTH_D;   // 768
    const int lane = threadIdx.x & 31;
    const _Float16* aprow = agg + (size_t)(row0 + (lane & 15)) * KTOT + (lane >> 4) * 8;
    v8f acc0 = {}; v8f acc1 = {};
#pragma unroll 4
    for (int c = 0; c < KTOT / 32; ++c) {
        if ((c + 1) * 32 < KTOT)
            __builtin_prefetch((const void*)(aprow + (c + 1) * 32), 0, 1);
        v16h a  = load_a_rowmajor(agg, row0, KTOT, c * 32);
        v16h b0 = load_b_ready(bWk, c * 2 + 0);
        v16h b1 = load_b_ready(bWk, c * 2 + 1);
        acc0 = wmma_f16(a, b0, acc0);
        acc1 = wmma_f16(a, b1, acc1);
    }
    store_d(acc0, [&](int m, int n, float v) {
        convL[(size_t)(row0 + m) * WIDTH_D + n] = (_Float16)lrelu_f(v, NEG_IN);
    });
    store_d(acc1, [&](int m, int n, float v) {
        convL[(size_t)(row0 + m) * WIDTH_D + 16 + n] = (_Float16)lrelu_f(v, NEG_IN);
    });
}

// ---------------------------------------------------------------------------
// k5: out = convL(N,32) @ W_out(32,128) + x   (f32 out, residual add)
// ---------------------------------------------------------------------------
__global__ __launch_bounds__(32)
void k5_output(const _Float16* __restrict__ convL, const _Float16* __restrict__ bWout,
               const float* __restrict__ x, float* __restrict__ out) {
    const int row0 = blockIdx.x * 16;
    v16h a = load_a_rowmajor(convL, row0, WIDTH_D, 0);
#pragma unroll
    for (int t = 0; t < C_DIM / 16; ++t) {
        v16h b = load_b_ready(bWout, t);
        v8f z = {};
        v8f d = wmma_f16(a, b, z);
        store_d(d, [&](int m, int n, float v) {
            const size_t idx = (size_t)(row0 + m) * C_DIM + t * 16 + n;
            out[idx] = v + x[idx];
        });
    }
}

// ---------------------------------------------------------------------------
// Host-side launcher
// ---------------------------------------------------------------------------
extern "C" void kernel_launch(void* const* d_in, const int* in_sizes, int n_in,
                              void* d_out, int out_size, void* d_ws, size_t ws_size,
                              hipStream_t stream) {
    const float* x     = (const float*)d_in[0];
    const float* pos   = (const float*)d_in[1];
    const float* seq   = (const float*)d_in[2];
    const float* ori   = (const float*)d_in[3];
    const float* W_in  = (const float*)d_in[4];
    const float* Ws0   = (const float*)d_in[5];
    const float* bs0   = (const float*)d_in[6];
    const float* Wk    = (const float*)d_in[7];
    const float* W_out = (const float*)d_in[8];
    const int*   src   = (const int*)d_in[9];
    const int*   dst   = (const int*)d_in[10];

    const int N   = in_sizes[0] / C_DIM;   // 8192
    const int E   = in_sizes[9];           // 139264
    const int epn = E / N;                 // 17 edges per node, grouped by dst

    // Workspace layout, 32B aligned chunks.
    char* ws = (char*)d_ws;
    size_t off = 0;
    auto alloc = [&](size_t bytes) {
        char* p = ws + off;
        off = (off + bytes + 31) & ~(size_t)31;
        return p;
    };
    _Float16* xr    = (_Float16*)alloc((size_t)N * C_DIM * 2);
    _Float16* h_f16 = (_Float16*)alloc((size_t)N * WIDTH_D * 2);
    _Float16* kern  = (_Float16*)alloc((size_t)E * KC_D * 2 + 64);  // pad for b128 tail
    _Float16* agg   = (_Float16*)alloc((size_t)N * KC_D * WIDTH_D * 2);
    _Float16* convL = (_Float16*)alloc((size_t)N * WIDTH_D * 2);
    _Float16* bWin  = (_Float16*)alloc((size_t)(C_DIM / 32) * 2 * 32 * 16 * 2);
    _Float16* bWk   = (_Float16*)alloc((size_t)(KC_D * WIDTH_D / 32) * 2 * 32 * 16 * 2);
    _Float16* bWout = (_Float16*)alloc((size_t)1 * 8 * 32 * 16 * 2);

    float* out = (float*)d_out;
    const int mtiles = N / 16;

    // Prep passes
    const int total4 = N * C_DIM / 4;
    k0_lrelu_x<<<(total4 + 255) / 256, 256, 0, stream>>>(x, xr, total4);
    k0_pack_b<<<(C_DIM / 32) * (WIDTH_D / 16), 32, 0, stream>>>(W_in, bWin, WIDTH_D);
    k0_pack_b<<<(KC_D * WIDTH_D / 32) * (WIDTH_D / 16), 32, 0, stream>>>(Wk, bWk, WIDTH_D);
    k0_pack_b<<<(WIDTH_D / 32) * (C_DIM / 16), 32, 0, stream>>>(W_out, bWout, C_DIM);

    // Main pipeline
    k1_input_mlp<<<mtiles, 32, 0, stream>>>(xr, bWin, h_f16);
    k2_edge_kern<<<(E + 255) / 256, 256, 0, stream>>>(pos, seq, ori, Ws0, bs0,
                                                      src, dst, E, kern);
    k3_aggregate<<<N, 32, 0, stream>>>(h_f16, kern, src, epn, agg);
    k4_conv_gemm<<<mtiles, 32, 0, stream>>>(agg, bWk, convL);
    k5_output<<<mtiles, 32, 0, stream>>>(convL, bWout, x, out);
}
